// CirculantMultiHeadAttention_1589137900082
// MI455X (gfx1250) — matrix operand order repaired and checked by reading, not compiled, not run on hardware
//
#include <hip/hip_runtime.h>
#include <hip/hip_bf16.h>

// CirculantMultiHeadAttention for MI455X (gfx1250, wave32, WMMA).
// Circulant blocks -> dense bf16 weights; all matmuls via v_wmma_f32_16x16x32_bf16.
// GEMM: 64x64 tile per wave (16 WMMAs / 16 b128 loads per K-chunk of 32).
// Requires ws_size >= 144 MiB.

typedef __attribute__((ext_vector_type(16))) __bf16 v16bf;
typedef __attribute__((ext_vector_type(8)))  float  v8f;

union Frag16 { v16bf v; uint4 q[2]; };   // 16 bf16 = 32B = two b128 loads

static __device__ __forceinline__ __bf16 f2bf(float f) {
    unsigned u = __builtin_bit_cast(unsigned, f);
    u += 0x7fffu + ((u >> 16) & 1u);               // round-to-nearest-even
    unsigned short h = (unsigned short)(u >> 16);
    return __builtin_bit_cast(__bf16, h);
}
static __device__ __forceinline__ float bf2f(__bf16 b) {
    unsigned short h = __builtin_bit_cast(unsigned short, b);
    unsigned u = ((unsigned)h) << 16;
    return __builtin_bit_cast(float, u);
}

static __device__ __forceinline__ v8f wmma_bf16(const Frag16& a, const Frag16& b, v8f c) {
    return __builtin_amdgcn_wmma_f32_16x16x32_bf16(
        false, a.v, false, b.v, (short)0, c, false, false);
}

// ---------------- circulant expansion: W_dense[n][k] = w[qb][pb][(i-j) mod bs]
__global__ void k_expand_w(const float* __restrict__ w, __bf16* __restrict__ Wd,
                           int D, int bs, int nb) {
    size_t idx = (size_t)blockIdx.x * blockDim.x + threadIdx.x;
    if (idx >= (size_t)D * D) return;
    int n = (int)(idx / D), k = (int)(idx % D);
    int qb = n / bs, i = n % bs;
    int pb = k / bs, j = k % bs;
    int g = i - j; if (g < 0) g += bs;
    Wd[idx] = f2bf(w[((size_t)qb * nb + pb) * bs + g]);
}

// ---------------- f32 -> bf16 elementwise
__global__ void k_to_bf16(const float* __restrict__ src, __bf16* __restrict__ dst, size_t n) {
    size_t i = (size_t)blockIdx.x * blockDim.x + threadIdx.x;
    if (i < n) dst[i] = f2bf(src[i]);
}

// ---------------- generic GEMM: C[m][n] = sum_k A[m][k]*W[n][k]
// A: bf16 row-major MxK, W: bf16 row-major NxK. One wave computes a 64x64 tile:
// 4 A-frags x 4 B-frags -> 16 WMMAs per K-chunk, each operand loaded once.
template <bool F32OUT>
__global__ __launch_bounds__(32)
void k_gemm(const __bf16* __restrict__ A, const __bf16* __restrict__ W,
            void* __restrict__ Cout, int M, int N, int K) {
    const int lane = threadIdx.x;
    const int mrow = lane & 15;
    const int hi   = lane >> 4;            // 0: lanes 0-15, 1: lanes 16-31
    const int m0 = blockIdx.y * 64;
    const int n0 = blockIdx.x * 64;

    const __bf16* aBase = A + (size_t)(m0 + mrow) * K + hi * 16;  // A-frag lane row
    const __bf16* wBase = W + (size_t)(n0 + mrow) * K + hi * 16;  // B-frag lane col

    v8f acc[4][4] = {};
    for (int k0 = 0; k0 < K; k0 += 32) {
        Frag16 af[4];
#pragma unroll
        for (int mf = 0; mf < 4; ++mf) {
            const __bf16* ar = aBase + (size_t)(mf * 16) * K + k0;
            af[mf].q[0] = *(const uint4*)(ar);        // k: hi*16 .. +7
            af[mf].q[1] = *(const uint4*)(ar + 8);    // k: hi*16+8 .. +15
        }
#pragma unroll
        for (int nf = 0; nf < 4; ++nf) {
            Frag16 bf;
            const __bf16* wr = wBase + (size_t)(nf * 16) * K + k0;
            bf.q[0] = *(const uint4*)(wr);            // k: hi*16 .. +7
            bf.q[1] = *(const uint4*)(wr + 8);        // k: hi*16+8 .. +15
#pragma unroll
            for (int mf = 0; mf < 4; ++mf)
                acc[mf][nf] = wmma_bf16(af[mf], bf, acc[mf][nf]);
        }
    }
#pragma unroll
    for (int mf = 0; mf < 4; ++mf)
#pragma unroll
        for (int nf = 0; nf < 4; ++nf)
#pragma unroll
            for (int r = 0; r < 8; ++r) {
                int m = m0 + mf * 16 + r + 8 * hi;   // C layout: vgpr=m%8, half=m>=8
                int n = n0 + nf * 16 + mrow;
                if (F32OUT) ((float*)Cout)[(size_t)m * N + n] = acc[mf][nf][r];
                else        ((__bf16*)Cout)[(size_t)m * N + n] = f2bf(acc[mf][nf][r]);
            }
}

// ---------------- RoPE + transpose (B,T,H,hd) -> (B*H, T, hd)
__global__ void k_rope(const __bf16* __restrict__ Xb, __bf16* __restrict__ Xh,
                       int B, int T, int H, int hd) {
    size_t idx = (size_t)blockIdx.x * blockDim.x + threadIdx.x;   // over B*T*H*(hd/2)
    size_t total = (size_t)B * T * H * (hd / 2);
    if (idx >= total) return;
    int i = (int)(idx % (hd / 2)); size_t rem = idx / (hd / 2);
    int h = (int)(rem % H); rem /= H;
    int t = (int)(rem % T); int b = (int)(rem / T);
    // inv = theta^(-2i/hd)
    float inv = __expf((-2.0f * i / (float)hd) * 9.210340371976184f); // ln(10000)
    float ang = (float)t * inv, c, s;
    __sincosf(ang, &s, &c);   // __sincosf(x, sin*, cos*)
    const __bf16* src = Xb + ((size_t)(b * T + t) * H + h) * hd;
    float xe = bf2f(src[2 * i]), xo = bf2f(src[2 * i + 1]);
    __bf16* dst = Xh + ((size_t)(b * H + h) * T + t) * hd;
    dst[2 * i]     = f2bf(xe * c - xo * s);
    dst[2 * i + 1] = f2bf(xe * s + xo * c);
}

// ---------------- V transpose (B,T,H,hd) -> (B*H, hd, T)
__global__ void k_vtrans(const __bf16* __restrict__ Vb, __bf16* __restrict__ Vt,
                         int B, int T, int H, int hd) {
    size_t idx = (size_t)blockIdx.x * blockDim.x + threadIdx.x;   // over B*T*H*hd
    size_t total = (size_t)B * T * H * hd;
    if (idx >= total) return;
    int d = (int)(idx % hd); size_t rem = idx / hd;
    int h = (int)(rem % H); rem /= H;
    int t = (int)(rem % T); int b = (int)(rem / T);
    Vt[((size_t)(b * H + h) * hd + d) * T + t] = Vb[idx];
}

// ---------------- flash attention: 1 wave per 16 query rows of one (b,h)
__global__ __launch_bounds__(32)
void k_flash(const __bf16* __restrict__ Qh, const __bf16* __restrict__ Kh,
             const __bf16* __restrict__ Vt, __bf16* __restrict__ Ab,
             int B, int T, int H, int hd /*128*/) {
    const int lane = threadIdx.x;
    const int mrow = lane & 15;
    const int hi   = lane >> 4;
    const int q0 = blockIdx.x * 16;
    const int bh = blockIdx.y;
    const int b = bh / H, h = bh % H;
    const float scale = 0.08838834764831845f;   // 1/sqrt(128)

    __shared__ float  S[16][32];
    __shared__ __bf16 P[16][32];
    __shared__ float rowM[16], rowL[16], rowScale[16];
    if (lane < 16) { rowM[lane] = -3.0e38f; rowL[lane] = 0.0f; }
    __syncthreads();

    // Q fragments: 4 feature-chunks of 32
    Frag16 qf[4];
    const __bf16* qbase = Qh + ((size_t)bh * T + q0 + mrow) * hd;
#pragma unroll
    for (int c = 0; c < 4; ++c) {
        qf[c].q[0] = *(const uint4*)(qbase + c * 32 + hi * 16);
        qf[c].q[1] = *(const uint4*)(qbase + c * 32 + hi * 16 + 8);
    }

    v8f acc[8] = {};                         // O tile: 16 x 128 f32
    const int kvend = q0 + 16;               // causal, exclusive
    for (int j0 = 0; j0 < kvend; j0 += 32) {
        // S = Q K^T for kv columns [j0, j0+32)
        v8f s0 = {}, s1 = {};
#pragma unroll
        for (int c = 0; c < 4; ++c) {
            Frag16 kf;
            const __bf16* kb = Kh + ((size_t)bh * T + j0 + mrow) * hd + c * 32 + hi * 16;
            kf.q[0] = *(const uint4*)(kb);
            kf.q[1] = *(const uint4*)(kb + 8);
            s0 = wmma_bf16(qf[c], kf, s0);
        }
#pragma unroll
        for (int c = 0; c < 4; ++c) {
            Frag16 kf;
            const __bf16* kb = Kh + ((size_t)bh * T + j0 + 16 + mrow) * hd + c * 32 + hi * 16;
            kf.q[0] = *(const uint4*)(kb);
            kf.q[1] = *(const uint4*)(kb + 8);
            s1 = wmma_bf16(qf[c], kf, s1);
        }
        // scale + causal mask, scatter C-fragments into LDS
#pragma unroll
        for (int r = 0; r < 8; ++r) {
            int m = r + 8 * hi;
            int qi = q0 + m;
            S[m][mrow]      = (j0 + mrow      <= qi) ? s0[r] * scale : -3.0e38f;
            S[m][16 + mrow] = (j0 + 16 + mrow <= qi) ? s1[r] * scale : -3.0e38f;
        }
        __syncthreads();
        // online softmax: one row per lane (lanes 0-15)
        if (lane < 16) {
            int r = lane;
            float mOld = rowM[r], mNew = mOld;
#pragma unroll
            for (int j = 0; j < 32; ++j) mNew = fmaxf(mNew, S[r][j]);
            float sum = 0.0f;
#pragma unroll
            for (int j = 0; j < 32; ++j) {
                float p = __expf(S[r][j] - mNew);
                P[r][j] = f2bf(p);
                sum += p;
            }
            float sc = __expf(mOld - mNew);
            rowL[r] = rowL[r] * sc + sum;
            rowM[r] = mNew;
            rowScale[r] = sc;
        }
        __syncthreads();
        // re-stripe P (C-layout -> A-layout) through LDS, rescale O, then O += P V
        Frag16 pf;
        {
            const __bf16* pb = &P[mrow][hi * 16];
            pf.q[0] = *(const uint4*)(pb);
            pf.q[1] = *(const uint4*)(pb + 8);
        }
        float scl[8];
#pragma unroll
        for (int r = 0; r < 8; ++r) scl[r] = rowScale[r + 8 * hi];
#pragma unroll
        for (int f = 0; f < 8; ++f)
#pragma unroll
            for (int r = 0; r < 8; ++r) acc[f][r] *= scl[r];
#pragma unroll
        for (int f = 0; f < 8; ++f) {
            Frag16 vf;   // B frag: lane = feature col, 16 contiguous kv positions
            const __bf16* vb = Vt + ((size_t)bh * hd + f * 16 + mrow) * T + j0 + hi * 16;
            vf.q[0] = *(const uint4*)(vb);
            vf.q[1] = *(const uint4*)(vb + 8);
            acc[f] = wmma_bf16(pf, vf, acc[f]);
        }
        __syncthreads();
    }
    // normalize, write attention output back in (B*T, D) layout for O-proj GEMM
    float invl[8];
#pragma unroll
    for (int r = 0; r < 8; ++r) invl[r] = 1.0f / rowL[r + 8 * hi];
#pragma unroll
    for (int f = 0; f < 8; ++f)
#pragma unroll
        for (int r = 0; r < 8; ++r) {
            int m = r + 8 * hi;
            int n = f * 16 + mrow;
            Ab[(size_t)(b * T + q0 + m) * (H * hd) + h * hd + n] = f2bf(acc[f][r] * invl[r]);
        }
}

extern "C" void kernel_launch(void* const* d_in, const int* in_sizes, int n_in,
                              void* d_out, int out_size, void* d_ws, size_t ws_size,
                              hipStream_t stream) {
    (void)in_sizes; (void)n_in; (void)out_size; (void)ws_size;
    const int B = 2, T = 2048, D = 2048, H = 16, hd = 128, bs = 512, nb = 4;
    const int M = B * T;                       // 4096 rows

    const float* x  = (const float*)d_in[0];
    const float* wq = (const float*)d_in[1];
    const float* wk = (const float*)d_in[2];
    const float* wv = (const float*)d_in[3];
    const float* wo = (const float*)d_in[4];
    // d_in[5] (mask) unused: causal mask applied analytically.
    float* out = (float*)d_out;

    char* ws = (char*)d_ws;
    const size_t MiB = 1024 * 1024;
    __bf16* Wq = (__bf16*)(ws + 0 * MiB);      //  8 MiB each (D*D bf16)
    __bf16* Wk = (__bf16*)(ws + 8 * MiB);
    __bf16* Wv = (__bf16*)(ws + 16 * MiB);
    __bf16* Wo = (__bf16*)(ws + 24 * MiB);
    __bf16* Xb = (__bf16*)(ws + 32 * MiB);     // 16 MiB (M*D bf16)
    __bf16* Qb = (__bf16*)(ws + 48 * MiB);
    __bf16* Kb = (__bf16*)(ws + 64 * MiB);
    __bf16* Vb = (__bf16*)(ws + 80 * MiB);
    __bf16* Qh = (__bf16*)(ws + 96 * MiB);
    __bf16* Kh = (__bf16*)(ws + 112 * MiB);
    __bf16* Vt = (__bf16*)(ws + 128 * MiB);
    __bf16* Ab = Qb;                           // Qb dead after RoPE -> reuse

    const size_t DD = (size_t)D * D;           // 4M
    const size_t MD = (size_t)M * D;           // 8M
    dim3 blk(256);

    // 1) expand circulant weights to dense bf16
    k_expand_w<<<dim3((DD + 255) / 256), blk, 0, stream>>>(wq, Wq, D, bs, nb);
    k_expand_w<<<dim3((DD + 255) / 256), blk, 0, stream>>>(wk, Wk, D, bs, nb);
    k_expand_w<<<dim3((DD + 255) / 256), blk, 0, stream>>>(wv, Wv, D, bs, nb);
    k_expand_w<<<dim3((DD + 255) / 256), blk, 0, stream>>>(wo, Wo, D, bs, nb);
    // 2) x -> bf16
    k_to_bf16<<<dim3((MD + 255) / 256), blk, 0, stream>>>(x, Xb, MD);

    // 3) QKV projections (WMMA GEMMs), 64x64 tile per wave
    dim3 gg(D / 64, M / 64);
    k_gemm<false><<<gg, dim3(32), 0, stream>>>(Xb, Wq, (void*)Qb, M, D, D);
    k_gemm<false><<<gg, dim3(32), 0, stream>>>(Xb, Wk, (void*)Kb, M, D, D);
    k_gemm<false><<<gg, dim3(32), 0, stream>>>(Xb, Wv, (void*)Vb, M, D, D);

    // 4) RoPE + layout transforms
    size_t nrope = (size_t)B * T * H * (hd / 2);
    k_rope<<<dim3((nrope + 255) / 256), blk, 0, stream>>>(Qb, Qh, B, T, H, hd);
    k_rope<<<dim3((nrope + 255) / 256), blk, 0, stream>>>(Kb, Kh, B, T, H, hd);
    k_vtrans<<<dim3((MD + 255) / 256), blk, 0, stream>>>(Vb, Vt, B, T, H, hd);

    // 5) causal flash attention
    k_flash<<<dim3(T / 16, B * H), dim3(32), 0, stream>>>(Qh, Kh, Vt, Ab, B, T, H, hd);

    // 6) output projection -> f32
    k_gemm<true><<<gg, dim3(32), 0, stream>>>(Ab, Wo, (void*)out, M, D, D);
}
